// SIFTDescriptor_37434934952192
// MI455X (gfx1250) — matrix operand order repaired and checked by reading, not compiled, not run on hardware
//
#include <hip/hip_runtime.h>

typedef __attribute__((ext_vector_type(16))) _Float16 v16h;
typedef __attribute__((ext_vector_type(8)))  float    v8f;

#define PATCH   41
#define NPIX    (PATCH * PATCH)
#define PPITCH  48
#define XPITCH  64
#define NBIN    8
#define TWO_PI  6.283185307179586f

union ABFrag { uint4 u[2]; v16h h; };
union S8     { uint4 u;    _Float16 f[8]; };

__device__ __forceinline__ float block_sum(float v, float* red, int tid) {
    red[tid] = v;
    __syncthreads();
    for (int s = 128; s > 0; s >>= 1) {
        if (tid < s) red[tid] += red[tid + s];
        __syncthreads();
    }
    float r = red[0];
    __syncthreads();
    return r;
}

__global__ __launch_bounds__(256) void sift_desc_kernel(
    const float* __restrict__ input, const float* __restrict__ gk,
    float* __restrict__ out, int npatch)
{
    // 7872 + 49152 + 2048 + 4096 + 512 + 1024 = 64704 B of LDS
    __shared__ __align__(16) float    patch[PATCH * PPITCH];
    __shared__ __align__(16) _Float16 masked[NBIN * 48 * XPITCH]; // [bin][y:48][x:64]
    __shared__ __align__(16) _Float16 Wbuf[16 * 64];              // [row n][k], rows>=4 zero
    __shared__ __align__(16) _Float16 Sbuf[NBIN * 4 * 64];        // [bin][ox][y:64]
    __shared__ __align__(16) float    hist[128];
    __shared__ __align__(16) float    red[256];

    const int tid = threadIdx.x;
    const int blk = blockIdx.x;
    if (blk >= npatch) return;

    const int wave = tid >> 5;
    const int lane = tid & 31;
    const int h    = lane >> 4;   // lane half: selects K sub-range of fragments
    const int mn   = lane & 15;   // row (A) / column (B) index within tile

    // ---- Phase 0: zero masked+S, load patch, build separable triangular W ----
    {
        const uint4 z = {0u, 0u, 0u, 0u};
        uint4* mz = (uint4*)masked;                 // 3072 x 16B
        for (int i = tid; i < 3072; i += 256) mz[i] = z;
        uint4* sz = (uint4*)Sbuf;                   // 256 x 16B
        for (int i = tid; i < 256; i += 256) sz[i] = z;

        const float* src = input + (size_t)blk * NPIX;
        for (int p = tid; p < NPIX; p += 256) {
            int y = p / PATCH, x = p - y * PATCH;
            patch[y * PPITCH + x] = src[p];
        }
        // W[n][x] = t[x - n*10 + 4] over 16-wide window, t[i] = (8 - |i+0.5-8|)/8
        for (int i = tid; i < 16 * 64; i += 256) {
            int r = i >> 6, x = i & 63;
            float v = 0.0f;
            if (r < 4 && x < PATCH) {
                int k = x - r * 10 + 4;
                if (k >= 0 && k < 16) {
                    float xc2 = 8.0f - fabsf((float)k + 0.5f - 8.0f);
                    v = xc2 * 0.125f;
                }
            }
            Wbuf[i] = (_Float16)v;
        }
    }
    __syncthreads();

    // ---- Phase 1: gradients -> gaussian-weighted mag, soft orientation bins ----
    for (int p = tid; p < NPIX; p += 256) {
        int y = p / PATCH, x = p - y * PATCH;
        float xl = patch[y * PPITCH + (x > 0 ? x - 1 : 0)];
        float xr = patch[y * PPITCH + (x < PATCH - 1 ? x + 1 : PATCH - 1)];
        float yu = patch[(y > 0 ? y - 1 : 0) * PPITCH + x];
        float yd = patch[(y < PATCH - 1 ? y + 1 : PATCH - 1) * PPITCH + x];
        float gx = 0.5f * (xr - xl);
        float gy = 0.5f * (yd - yu);
        // x1024: cancels in L2-normalization, keeps f16 in normal range
        float mag = sqrtf(gx * gx + gy * gy + 1e-10f) * gk[p] * 1024.0f;
        float ori = atan2f(gy, gx + 1e-10f) + TWO_PI;
        float obig = ori * ((float)NBIN / TWO_PI);
        float b0f = floorf(obig);
        float w1 = obig - b0f;
        int b0 = ((int)b0f) & 7;
        int b1 = (b0 + 1) & 7;
        masked[(b0 * 48 + y) * XPITCH + x] = (_Float16)((1.0f - w1) * mag);
        masked[(b1 * 48 + y) * XPITCH + x] = (_Float16)(w1 * mag);
    }
    __syncthreads();

    // ---- Phase 2: x-pooling via WMMA: S_b[y,ox] = sum_x masked_b[y,x]*W[ox][x] ----
    for (int j = wave; j < 24; j += 8) {          // 8 bins x 3 M-tiles
        const int b  = j / 3;
        const int mt = j % 3;
        v8f acc = {0.f, 0.f, 0.f, 0.f, 0.f, 0.f, 0.f, 0.f};
        const _Float16* arow = &masked[(b * 48 + mt * 16 + mn) * XPITCH];
        const _Float16* brow = &Wbuf[mn * 64];
#pragma unroll
        for (int kk = 0; kk < 2; ++kk) {
            const int kb = kk * 32;
            ABFrag A, Bf;
            A.u[0]  = *(const uint4*)(arow + kb + h * 8);
            A.u[1]  = *(const uint4*)(arow + kb + 16 + h * 8);
            Bf.u[0] = *(const uint4*)(brow + kb + h * 8);
            Bf.u[1] = *(const uint4*)(brow + kb + 16 + h * 8);
            acc = __builtin_amdgcn_wmma_f32_16x16x32_f16(
                false, A.h, false, Bf.h, (short)0, acc, false, false);
        }
        if (mn < 4) {  // D[M=v+8h, n=ox] -> S[b][ox][y-run of 8]
            S8 s;
#pragma unroll
            for (int v = 0; v < 8; ++v) s.f[v] = (_Float16)acc[v];
            *(uint4*)&Sbuf[b * 256 + mn * 64 + mt * 16 + h * 8] = s.u;
        }
    }
    __syncthreads();

    // ---- Phase 3: y-pooling via WMMA: pooled[b,oy,ox] = sum_y S_b[y,ox]*W[oy][y] ----
    if (wave < 2) {                                // rows r = (b%4)*4+ox, 2 tiles
        const int mt2 = wave;
        const int b   = mt2 * 4 + (mn >> 2);
        const int ox  = mn & 3;
        v8f acc = {0.f, 0.f, 0.f, 0.f, 0.f, 0.f, 0.f, 0.f};
        const _Float16* arow = &Sbuf[b * 256 + ox * 64];
        const _Float16* brow = &Wbuf[mn * 64];
#pragma unroll
        for (int kk = 0; kk < 2; ++kk) {
            const int kb = kk * 32;
            ABFrag A, Bf;
            A.u[0]  = *(const uint4*)(arow + kb + h * 8);
            A.u[1]  = *(const uint4*)(arow + kb + 16 + h * 8);
            Bf.u[0] = *(const uint4*)(brow + kb + h * 8);
            Bf.u[1] = *(const uint4*)(brow + kb + 16 + h * 8);
            acc = __builtin_amdgcn_wmma_f32_16x16x32_f16(
                false, A.h, false, Bf.h, (short)0, acc, false, false);
        }
        if (mn < 4) {  // D[M=v+8h, n=oy] -> hist[b*16 + oy*4 + ox]
#pragma unroll
            for (int v = 0; v < 8; ++v) {
                const int rr  = 8 * h + v;
                const int bb  = mt2 * 4 + (rr >> 2);
                const int oxx = rr & 3;
                hist[bb * 16 + mn * 4 + oxx] = acc[v];
            }
        }
    }
    __syncthreads();

    // ---- Phase 4: l2n -> clip(0.2) -> l2n -> rootsift ----
    float v0 = (tid < 128) ? hist[tid] : 0.0f;
    float ss = block_sum(v0 * v0, red, tid);
    float n1 = fmaxf(sqrtf(ss), 1e-12f);
    float a  = fminf(fmaxf(v0 / n1, 0.0f), 0.2f);
    float ss2 = block_sum(a * a, red, tid);
    float n2 = fmaxf(sqrtf(ss2), 1e-12f);
    float a2 = a / n2;
    float l1 = block_sum(fabsf(a2), red, tid);
    l1 = fmaxf(l1, 1e-12f);
    if (tid < 128) out[(size_t)blk * 128 + tid] = sqrtf(a2 / l1 + 1e-10f);
}

extern "C" void kernel_launch(void* const* d_in, const int* in_sizes, int n_in,
                              void* d_out, int out_size, void* d_ws, size_t ws_size,
                              hipStream_t stream) {
    (void)n_in; (void)out_size; (void)d_ws; (void)ws_size;
    const float* input = (const float*)d_in[0];
    const float* gk    = (const float*)d_in[1];
    // d_in[2] (pk) is unused: pk is separable and W is derived analytically.
    float* out = (float*)d_out;
    const int npatch = in_sizes[0] / NPIX;
    sift_desc_kernel<<<npatch, 256, 0, stream>>>(input, gk, out, npatch);
}
